// MultiHeadAttnLSTMModel_54090818126561
// MI455X (gfx1250) — compile-verified, matrix operand order checked by reading, not run on hardware
//
#include <hip/hip_runtime.h>
#include <hip/hip_bf16.h>

// ---------------- constants ----------------
#define SEQ   100
#define Dm    500
#define Dp    512
#define Hh    8
#define DKv   64
#define Bsz   256
#define G4D   2000
#define DIM2  100
#define SEQpA 112   // padded M/N for 100
#define SEQpK 128   // padded K for 100
#define MROWS 25600 // B*SEQ
#define RSLK  25616 // padded rows (+16 slack)

typedef __attribute__((ext_vector_type(16))) __bf16 bf16x16;
typedef __attribute__((ext_vector_type(8)))  float  f32x8;
typedef unsigned int u32x4 __attribute__((ext_vector_type(4)));
typedef int          i32x4v __attribute__((ext_vector_type(4)));
typedef int          i32x8v __attribute__((ext_vector_type(8)));

#if __has_builtin(__builtin_amdgcn_tensor_load_to_lds)
#define HAVE_TDM 1
#endif

__device__ __forceinline__ __bf16 f2bf(float f) {
  unsigned u = __float_as_uint(f);
  u += 0x7FFFu + ((u >> 16) & 1u);           // round-to-nearest-even
  unsigned short h = (unsigned short)(u >> 16);
  __bf16 r;
  __builtin_memcpy(&r, &h, sizeof(r));
  return r;
}

__device__ __forceinline__ float wave_sum(float v) {
  #pragma unroll
  for (int o = 16; o; o >>= 1) v += __shfl_xor(v, o, 32);
  return v;
}
__device__ __forceinline__ float wave_max(float v) {
  #pragma unroll
  for (int o = 16; o; o >>= 1) v = fmaxf(v, __shfl_xor(v, o, 32));
  return v;
}
__device__ __forceinline__ float sigm(float x) { return 1.0f / (1.0f + expf(-x)); }

// ---------------- generic batched bf16 WMMA GEMM ----------------
// C[M,N] = scale * (A[M,K] x Bt[N,K]^T) (+ bias[n]) (+ Add[row*ldadd+col])
// Block = 4 waves, one M-tile per block (A strip staged in LDS via TDM),
// each wave computes NT adjacent 16x16 N-tiles (register blocking).
#define GF_BIAS  1
#define GF_TRANS 2
#define GF_BF16  4
#define GF_ADD   8

template<int NT>
__global__ void gemm_bf16_kernel(
    const __bf16* __restrict__ A,  long sAy, long sAz,
    const __bf16* __restrict__ Bt, long sBy, long sBz,
    void* __restrict__ Cv,         long sCy, long sCz,   // BYTE strides for C
    const float* __restrict__ bias,
    const float* __restrict__ Add, int ldadd,
    int M, int N, int K, int lda, int ldb, int ldc,
    int tilesN, int nChunks, int flags, float scale)
{
  __shared__ __bf16 Asmem[16 * 512];                     // K <= 512 always

  int lane  = threadIdx.x & 31;
  int wid   = threadIdx.x >> 5;
  int tm    = blockIdx.x / nChunks;
  int chunk = blockIdx.x % nChunks;

  const __bf16* Ab = A  + (long)blockIdx.y * sAy + (long)blockIdx.z * sAz;
  const __bf16* Bb = Bt + (long)blockIdx.y * sBy + (long)blockIdx.z * sBz;
  char*         Cb = (char*)Cv + (long)blockIdx.y * sCy + (long)blockIdx.z * sCz;

  const __bf16* Astrip = Ab + (long)tm * 16 * lda;

  // ---- stage A strip (16 rows x K) into LDS ----
#ifdef HAVE_TDM
  if (wid == 0) {
    // Tensor DMA descriptor (D#), 2D tile: tile_dim0=K elems, tile_dim1=16 rows
    unsigned long long ga = (unsigned long long)(size_t)Astrip;
    unsigned lds = (unsigned)(size_t)(&Asmem[0]);        // addr[31:0] == LDS offset
    unsigned td0 = (unsigned)K;
    u32x4 g0;
    g0[0] = 1u;                                          // count=1, user D#
    g0[1] = lds;                                         // lds_addr
    g0[2] = (unsigned)ga;                                // global_addr[31:0]
    g0[3] = ((unsigned)(ga >> 32) & 0x01FFFFFFu) | (2u << 30); // addr[56:32], type=2
    i32x8v g1;
    g1[0] = 1 << 16;                                     // data_size=1 (2 bytes)
    g1[1] = (int)((td0 & 0xFFFFu) << 16);                // tensor_dim0[15:0]
    g1[2] = (int)((td0 >> 16) | (16u << 16));            // tensor_dim0[31:16], tensor_dim1=16
    g1[3] = (int)(td0 << 16);                            // tile_dim0 = K
    g1[4] = 16;                                          // tile_dim1=16, tile_dim2=0
    g1[5] = (int)(unsigned)lda;                          // tensor_dim0_stride[31:0]
    g1[6] = 0;
    g1[7] = 0;
    i32x4v gz4 = {0, 0, 0, 0};
    i32x8v gz8 = {0, 0, 0, 0, 0, 0, 0, 0};
    __builtin_amdgcn_tensor_load_to_lds(g0, g1, gz4, gz4, gz8, 0);
#if __has_builtin(__builtin_amdgcn_s_wait_tensorcnt)
    __builtin_amdgcn_s_wait_tensorcnt((short)0);
#endif
  }
#else
  {
    int cpr = K >> 3;                                    // 8-elem (16B) chunks per row
    for (int idx = threadIdx.x; idx < 16 * cpr; idx += 128) {
      int r = idx / cpr, c = idx % cpr;
      *(uint4*)&Asmem[r * K + c * 8] = *(const uint4*)&Astrip[(long)r * lda + c * 8];
    }
  }
#endif
  __syncthreads();

  int m = lane & 15;
  int g = lane >> 4;
  const __bf16* arl = &Asmem[m * K + g * 8];

  const __bf16* brp[NT];
  int tnv[NT];
  #pragma unroll
  for (int j = 0; j < NT; ++j) {
    int tn = chunk * (4 * NT) + wid * NT + j;
    tnv[j] = tn;
    int tnc = (tn < tilesN) ? tn : (tilesN - 1);
    brp[j] = Bb + (long)(tnc * 16 + m) * ldb + g * 8;
  }

  const f32x8 zero = {0.f, 0.f, 0.f, 0.f, 0.f, 0.f, 0.f, 0.f};
  f32x8 acc[NT];
  #pragma unroll
  for (int j = 0; j < NT; ++j) acc[j] = zero;

  for (int k0 = 0; k0 < K; k0 += 32) {
    union { bf16x16 v; uint4 u[2]; } ua;
    ua.u[0] = *(const uint4*)(arl + k0);                 // ds_load_b128 from LDS
    ua.u[1] = *(const uint4*)(arl + k0 + 16);
    #pragma unroll
    for (int j = 0; j < NT; ++j) {
      union { bf16x16 v; uint4 u[2]; } ub;
      ub.u[0] = *(const uint4*)(brp[j] + k0);
      ub.u[1] = *(const uint4*)(brp[j] + k0 + 16);
      if (k0 + 32 < K) __builtin_prefetch(brp[j] + k0 + 32, 0, 3);
      acc[j] = __builtin_amdgcn_wmma_f32_16x16x32_bf16(
          false, ua.v, false, ub.v, (short)0, acc[j], false, false);
    }
  }

  int nlane = lane & 15;
  #pragma unroll
  for (int j = 0; j < NT; ++j) {
    int tn = tnv[j];
    if (tn >= tilesN) continue;                          // wave-uniform
    int col = tn * 16 + nlane;
    float bv = ((flags & GF_BIAS) && col < N) ? bias[col] : 0.f;
    #pragma unroll
    for (int r = 0; r < 8; ++r) {
      int row = tm * 16 + r + 8 * g;                     // C layout: VGPR r -> M = r + 8*(lane>=16)
      if (row >= M || col >= N) continue;
      float v = acc[j][r] * scale + bv;
      if (flags & GF_ADD) v += Add[(long)row * ldadd + col];
      if (flags & GF_TRANS) {
        ((float*)Cb)[(long)col * ldc + row] = v;
      } else if (flags & GF_BF16) {
        ((__bf16*)Cb)[(long)row * ldc + col] = f2bf(v);
      } else {
        ((float*)Cb)[(long)row * ldc + col] = v;
      }
    }
  }
}

// ---------------- conversion / fill kernels ----------------
__global__ void split_cvt_kernel(const float* __restrict__ din,
                                 __bf16* __restrict__ sysb, __bf16* __restrict__ refb)
{
  long i = (long)blockIdx.x * blockDim.x + threadIdx.x;
  if (i >= (long)MROWS * Dp) return;
  long row = i >> 9;
  int  d   = (int)(i & 511);
  int  b   = (int)(row / SEQ);
  int  s   = (int)(row % SEQ);
  long base = ((long)b * (2 * SEQ) + s) * Dm;
  float vs = (d < Dm) ? din[base + d] : 0.f;
  float vr = (d < Dm) ? din[base + (long)SEQ * Dm + d] : 0.f;
  sysb[i] = f2bf(vs);
  refb[i] = f2bf(vr);
}

__global__ void cvt_qkvw_kernel(const float* wq, const float* wk, const float* wv,
                                __bf16* qT, __bf16* kT, __bf16* vT)
{
  long i = (long)blockIdx.x * blockDim.x + threadIdx.x;
  if (i >= (long)Hh * DKv * Dp) return;
  int h = (int)(i / (DKv * Dp));
  int r = (int)(i % (DKv * Dp));
  int n = r / Dp;
  int d = r % Dp;
  long src = (long)h * Dm * DKv + (long)d * DKv + n;
  float a = (d < Dm) ? wq[src] : 0.f;
  float b = (d < Dm) ? wk[src] : 0.f;
  float c = (d < Dm) ? wv[src] : 0.f;
  qT[i] = f2bf(a); kT[i] = f2bf(b); vT[i] = f2bf(c);
}

// W is K x srcN (row-major); dst[n*Kp+d] = W[d*srcN+n]
__global__ void cvt_transpose_kernel(const float* __restrict__ W, __bf16* __restrict__ dst,
                                     int Nrows, int Nvalid, int Kp, int Kvalid, int srcN)
{
  long i = (long)blockIdx.x * blockDim.x + threadIdx.x;
  if (i >= (long)Nrows * Kp) return;
  int n = (int)(i / Kp);
  int d = (int)(i % Kp);
  float v = (n < Nvalid && d < Kvalid) ? W[(long)d * srcN + n] : 0.f;
  dst[i] = f2bf(v);
}

// W is N x Ksrc row-major; dst[n*Kp+d] = W[n*Ksrc+d]
__global__ void cvt_direct_kernel(const float* __restrict__ W, __bf16* __restrict__ dst,
                                  int Nrows, int Nvalid, int Kp, int Ksrc)
{
  long i = (long)blockIdx.x * blockDim.x + threadIdx.x;
  if (i >= (long)Nrows * Kp) return;
  int n = (int)(i / Kp);
  int d = (int)(i % Kp);
  float v = (n < Nvalid && d < Ksrc) ? W[(long)n * Ksrc + d] : 0.f;
  dst[i] = f2bf(v);
}

__global__ void vecadd_kernel(const float* a, const float* b, float* o, int n)
{
  int i = blockIdx.x * blockDim.x + threadIdx.x;
  if (i < n) o[i] = a[i] + b[i];
}

__global__ void fill_f32_kernel(float* p, long n)
{
  long i = (long)blockIdx.x * blockDim.x + threadIdx.x;
  if (i < n) p[i] = 0.f;
}
__global__ void fill_bf16_kernel(__bf16* p, long n)
{
  long i = (long)blockIdx.x * blockDim.x + threadIdx.x;
  if (i < n) p[i] = f2bf(0.f);
}

// ---------------- softmax (one wave per score row) ----------------
__global__ void softmax_kernel(const float* __restrict__ scores, __bf16* __restrict__ attn)
{
  int wid  = threadIdx.x >> 5;
  int lane = threadIdx.x & 31;
  long wrow = (long)blockIdx.x * 4 + wid;
  long total = (long)Bsz * Hh * SEQ;
  if (wrow >= total) return;
  long bh = wrow / SEQ;
  int  r  = (int)(wrow % SEQ);
  const float* src = scores + (bh * SEQpA + r) * SEQpA;
  __bf16*      dst = attn   + (bh * SEQpA + r) * SEQpK;

  float v[4];
  #pragma unroll
  for (int i = 0; i < 4; ++i) {
    int s = lane + 32 * i;
    v[i] = (s < SEQ) ? src[s] : -3.0e38f;
  }
  float mx = fmaxf(fmaxf(v[0], v[1]), fmaxf(v[2], v[3]));
  mx = wave_max(mx);
  float e[4], sum = 0.f;
  #pragma unroll
  for (int i = 0; i < 4; ++i) {
    int s = lane + 32 * i;
    e[i] = (s < SEQ) ? expf(v[i] - mx) : 0.f;
    sum += e[i];
  }
  sum = wave_sum(sum);
  float inv = 1.0f / sum;
  #pragma unroll
  for (int i = 0; i < 4; ++i) {
    int s = lane + 32 * i;
    dst[s] = f2bf((s < SEQ) ? e[i] * inv : 0.f);   // zero K-padding 100..127
  }
}

// ---------------- residual + layernorm (one wave per row) ----------------
__global__ void ln_kernel(const float* __restrict__ proj, const float* __restrict__ din,
                          const float* __restrict__ lg, const float* __restrict__ lb,
                          __bf16* __restrict__ xln)
{
  int wid  = threadIdx.x >> 5;
  int lane = threadIdx.x & 31;
  long row = (long)blockIdx.x * 4 + wid;
  if (row >= MROWS) return;
  int b = (int)(row / SEQ);
  int s = (int)(row % SEQ);
  const float* pr = proj + row * Dm;
  const float* rr = din + ((long)b * (2 * SEQ) + SEQ + s) * Dm;

  float x[16];
  float sum = 0.f, sq = 0.f;
  #pragma unroll
  for (int i = 0; i < 16; ++i) {
    int d = lane + 32 * i;
    float v = (d < Dm) ? (pr[d] + rr[d]) : 0.f;
    x[i] = v; sum += v; sq += v * v;
  }
  sum = wave_sum(sum);
  sq  = wave_sum(sq);
  float mu   = sum / (float)Dm;
  float var  = sq / (float)Dm - mu * mu;
  float rstd = rsqrtf(var + 1e-5f);
  __bf16* dst = xln + row * Dp;
  #pragma unroll
  for (int i = 0; i < 16; ++i) {
    int d = lane + 32 * i;
    float v = (d < Dm) ? ((x[i] - mu) * rstd * lg[d] + lb[d]) : 0.f;
    dst[d] = f2bf(v);
  }
}

// ---------------- LSTM gate pointwise ----------------
__global__ void lstm_point_kernel(const float* __restrict__ gates, float* __restrict__ c,
                                  __bf16* __restrict__ hbf,
                                  __bf16* __restrict__ ybf, float* __restrict__ yf,
                                  int t, int flags)
{
  long idx = (long)blockIdx.x * blockDim.x + threadIdx.x;
  if (idx >= (long)Bsz * Dm) return;
  int b = (int)(idx / Dm);
  int j = (int)(idx % Dm);
  const float* gr = gates + (long)b * G4D;
  float ig = sigm(gr[j]);
  float fg = sigm(gr[Dm + j]);
  float gg = tanhf(gr[2 * Dm + j]);
  float og = sigm(gr[3 * Dm + j]);
  float cn = fg * c[idx] + ig * gg;
  c[idx] = cn;
  float h = og * tanhf(cn);
  hbf[(long)b * Dp + j] = f2bf(h);
  if (flags & 1) ybf[((long)b * SEQ + t) * Dp + j] = f2bf(h);
  if (flags & 2) yf[((long)b * SEQ + t) * Dm + j] = h;
}

// ---------------- sequence-length detection ----------------
__global__ void lengths_kernel(const float* __restrict__ din, int* __restrict__ len)
{
  __shared__ int mi;
  if (threadIdx.x == 0) mi = SEQ;
  __syncthreads();
  int b = blockIdx.x;
  for (int s = threadIdx.x; s < SEQ; s += blockDim.x) {
    const float* p = din + ((long)b * (2 * SEQ) + SEQ + s) * Dm;
    float sum = 0.f;
    for (int d = 0; d < Dm; ++d) sum += fabsf(p[d]);
    if (ceilf(sum) == 0.0f) atomicMin(&mi, s);
  }
  __syncthreads();
  if (threadIdx.x == 0) {
    int L = mi - 1;
    if (L < 0) L = 0;
    len[b] = L;
  }
}

__global__ void sel_kernel(const float* __restrict__ ys2, const int* __restrict__ len,
                           __bf16* __restrict__ sel)
{
  long i = (long)blockIdx.x * blockDim.x + threadIdx.x;
  if (i >= (long)Bsz * Dp) return;
  int b = (int)(i >> 9);
  int d = (int)(i & 511);
  int L = len[b];
  float v = (d < Dm) ? ys2[((long)b * SEQ + L) * Dm + d] : 0.f;
  sel[i] = f2bf(v);
}

// ---------------- batchnorm stats / bn + leaky + fc3 ----------------
__global__ void bnstats_kernel(const float* __restrict__ y, float* mean, float* var)
{
  __shared__ float s1[256], s2[256];
  int j = blockIdx.x;
  int b = threadIdx.x;
  float v = y[(long)b * DIM2 + j];
  s1[b] = v; s2[b] = v * v;
  __syncthreads();
  for (int o = 128; o; o >>= 1) {
    if (b < o) { s1[b] += s1[b + o]; s2[b] += s2[b + o]; }
    __syncthreads();
  }
  if (b == 0) {
    float m = s1[0] / (float)Bsz;
    mean[j] = m;
    var[j]  = s2[0] / (float)Bsz - m * m;
  }
}

__global__ void bnfc3_kernel(const float* __restrict__ y, const float* mean, const float* var,
                             const float* g, const float* be,
                             const float* w3, const float* b3, float* __restrict__ out)
{
  __shared__ float sh[128];
  int b = blockIdx.x;
  int j = threadIdx.x;
  float acc = 0.f;
  if (j < DIM2) {
    float v = y[(long)b * DIM2 + j];
    v = (v - mean[j]) * rsqrtf(var[j] + 1e-5f) * g[j] + be[j];
    v = (v >= 0.f) ? v : 0.01f * v;
    acc = v * w3[j];
  }
  sh[j] = acc;
  __syncthreads();
  for (int o = 64; o; o >>= 1) {
    if (j < o) sh[j] += sh[j + o];
    __syncthreads();
  }
  if (j == 0) out[b] = sh[0] + b3[0];
}

// ---------------- host side ----------------
static void launch_gemm(hipStream_t st, int NT,
    const __bf16* A, long sAy, long sAz,
    const __bf16* Bt, long sBy, long sBz,
    void* C, long sCyB, long sCzB,
    const float* bias, const float* Add, int ldadd,
    int M, int N, int K, int lda, int ldb, int ldc,
    int Y, int Z, int flags, float scale)
{
  int tilesM = (M + 15) / 16;
  int tilesN = (N + 15) / 16;
  int per = 4 * NT;
  int nChunks = (tilesN + per - 1) / per;
  dim3 grid(tilesM * nChunks, Y, Z);
  if (NT == 1) {
    gemm_bf16_kernel<1><<<grid, dim3(128), 0, st>>>(A, sAy, sAz, Bt, sBy, sBz, C, sCyB, sCzB,
        bias, Add, ldadd, M, N, K, lda, ldb, ldc, tilesN, nChunks, flags, scale);
  } else if (NT == 2) {
    gemm_bf16_kernel<2><<<grid, dim3(128), 0, st>>>(A, sAy, sAz, Bt, sBy, sBz, C, sCyB, sCzB,
        bias, Add, ldadd, M, N, K, lda, ldb, ldc, tilesN, nChunks, flags, scale);
  } else {
    gemm_bf16_kernel<4><<<grid, dim3(128), 0, st>>>(A, sAy, sAz, Bt, sBy, sBz, C, sCyB, sCzB,
        bias, Add, ldadd, M, N, K, lda, ldb, ldc, tilesN, nChunks, flags, scale);
  }
}

extern "C" void kernel_launch(void* const* d_in, const int* in_sizes, int n_in,
                              void* d_out, int out_size, void* d_ws, size_t ws_size,
                              hipStream_t stream)
{
  (void)in_sizes; (void)n_in; (void)out_size; (void)ws_size;
  const float* data_in = (const float*)d_in[0];
  const float* wq    = (const float*)d_in[1];
  const float* wk    = (const float*)d_in[2];
  const float* wv    = (const float*)d_in[3];
  const float* wo    = (const float*)d_in[4];
  const float* wo_b  = (const float*)d_in[5];
  const float* ln_g  = (const float*)d_in[6];
  const float* ln_b  = (const float*)d_in[7];
  const float* Wih0  = (const float*)d_in[8];
  const float* Whh0  = (const float*)d_in[9];
  const float* bih0  = (const float*)d_in[10];
  const float* bhh0  = (const float*)d_in[11];
  const float* Wih1  = (const float*)d_in[12];
  const float* Whh1  = (const float*)d_in[13];
  const float* bih1  = (const float*)d_in[14];
  const float* bhh1  = (const float*)d_in[15];
  const float* fc1_w = (const float*)d_in[16];
  const float* fc1_b = (const float*)d_in[17];
  const float* bn_g  = (const float*)d_in[18];
  const float* bn_b  = (const float*)d_in[19];
  const float* fc3_w = (const float*)d_in[20];
  const float* fc3_b = (const float*)d_in[21];
  float* out = (float*)d_out;

  // ---- workspace carve-up ----
  char* ws = (char*)d_ws;
  size_t off = 0;
  auto take = [&](size_t bytes) -> char* {
    char* p = ws + off;
    off += (bytes + 255) & ~(size_t)255;
    return p;
  };
  __bf16* sys_bf = (__bf16*)take((size_t)RSLK * Dp * 2);
  __bf16* ref_bf = (__bf16*)take((size_t)RSLK * Dp * 2);
  __bf16* wqT    = (__bf16*)take((size_t)Hh * DKv * Dp * 2);
  __bf16* wkT    = (__bf16*)take((size_t)Hh * DKv * Dp * 2);
  __bf16* wvT    = (__bf16*)take((size_t)Hh * DKv * Dp * 2);
  __bf16* woT    = (__bf16*)take((size_t)Dp * Dp * 2);
  __bf16* Wih0b  = (__bf16*)take((size_t)G4D * Dp * 2);
  __bf16* Whh0b  = (__bf16*)take((size_t)G4D * Dp * 2);
  __bf16* Wih1b  = (__bf16*)take((size_t)G4D * Dp * 2);
  __bf16* Whh1b  = (__bf16*)take((size_t)G4D * Dp * 2);
  __bf16* fc1b   = (__bf16*)take((size_t)SEQpA * Dp * 2);
  float*  bsum0  = (float*)take(G4D * 4);
  float*  bsum1  = (float*)take(G4D * 4);
  __bf16* qb     = (__bf16*)take((size_t)Bsz * Hh * SEQpA * DKv * 2);
  __bf16* kb     = (__bf16*)take((size_t)Bsz * Hh * SEQpA * DKv * 2);
  __bf16* vTb    = (__bf16*)take((size_t)Bsz * Hh * DKv * SEQpK * 2);
  float*  scoresb= (float*)take((size_t)Bsz * Hh * SEQpA * SEQpA * 4);
  __bf16* attnb  = (__bf16*)take((size_t)Bsz * Hh * SEQpA * SEQpK * 2);
  __bf16* ctxb   = (__bf16*)take((size_t)RSLK * Dp * 2);
  float*  projb  = (float*)take((size_t)MROWS * Dm * 4);
  __bf16* xlnb   = (__bf16*)take((size_t)RSLK * Dp * 2);
  float*  preb   = (float*)take((size_t)MROWS * G4D * 4);
  float*  gatesb = (float*)take((size_t)Bsz * G4D * 4);
  __bf16* hbf    = (__bf16*)take((size_t)Bsz * Dp * 2);
  float*  cbuf   = (float*)take((size_t)Bsz * Dm * 4);
  __bf16* ys1b   = (__bf16*)take((size_t)RSLK * Dp * 2);
  float*  ys2b   = (float*)take((size_t)Bsz * SEQ * Dm * 4);
  int*    lenb   = (int*)take(Bsz * 4);
  __bf16* selb   = (__bf16*)take((size_t)Bsz * Dp * 2);
  float*  yb     = (float*)take((size_t)Bsz * DIM2 * 4);
  float*  bnm    = (float*)take(DIM2 * 4);
  float*  bnv    = (float*)take(DIM2 * 4);

  // ---- 1. weight conversions (f32 -> bf16, Bt = NxK layout, K padded) ----
  {
    long n = (long)Hh * DKv * Dp;
    cvt_qkvw_kernel<<<(unsigned)((n + 255) / 256), 256, 0, stream>>>(wq, wk, wv, wqT, wkT, wvT);
  }
  cvt_transpose_kernel<<<(Dp * Dp + 255) / 256, 256, 0, stream>>>(wo, woT, Dp, Dm, Dp, Hh * DKv, Dm);
  {
    long n = (long)G4D * Dp;
    unsigned g = (unsigned)((n + 255) / 256);
    cvt_direct_kernel<<<g, 256, 0, stream>>>(Wih0, Wih0b, G4D, G4D, Dp, Dm);
    cvt_direct_kernel<<<g, 256, 0, stream>>>(Whh0, Whh0b, G4D, G4D, Dp, Dm);
    cvt_direct_kernel<<<g, 256, 0, stream>>>(Wih1, Wih1b, G4D, G4D, Dp, Dm);
    cvt_direct_kernel<<<g, 256, 0, stream>>>(Whh1, Whh1b, G4D, G4D, Dp, Dm);
  }
  cvt_direct_kernel<<<(SEQpA * Dp + 255) / 256, 256, 0, stream>>>(fc1_w, fc1b, SEQpA, DIM2, Dp, Dm);
  vecadd_kernel<<<(G4D + 255) / 256, 256, 0, stream>>>(bih0, bhh0, bsum0, G4D);
  vecadd_kernel<<<(G4D + 255) / 256, 256, 0, stream>>>(bih1, bhh1, bsum1, G4D);

  // ---- 2. activation conversion, zero vT pad ----
  {
    long n = (long)MROWS * Dp;
    split_cvt_kernel<<<(unsigned)((n + 255) / 256), 256, 0, stream>>>(data_in, sys_bf, ref_bf);
  }
  {
    long n = (long)Bsz * Hh * DKv * SEQpK;
    fill_bf16_kernel<<<(unsigned)((n + 255) / 256), 256, 0, stream>>>(vTb, n);
  }

  // ---- 3. attention QKV ----
  launch_gemm(stream, 1, ref_bf, (long)SEQ * Dp, 0, wqT, 0, (long)DKv * Dp,
              qb, (long)Hh * SEQpA * DKv * 2, (long)SEQpA * DKv * 2,
              nullptr, nullptr, 0, SEQ, DKv, Dp, Dp, Dp, DKv, Bsz, Hh, 0, 1.f);
  launch_gemm(stream, 1, sys_bf, (long)SEQ * Dp, 0, wkT, 0, (long)DKv * Dp,
              kb, (long)Hh * SEQpA * DKv * 2, (long)SEQpA * DKv * 2,
              nullptr, nullptr, 0, SEQ, DKv, Dp, Dp, Dp, DKv, Bsz, Hh, 0, 1.f);
  launch_gemm(stream, 1, sys_bf, (long)SEQ * Dp, 0, wvT, 0, (long)DKv * Dp,
              vTb, (long)Hh * DKv * SEQpK * 2, (long)DKv * SEQpK * 2,
              nullptr, nullptr, 0, SEQ, DKv, Dp, Dp, Dp, SEQpK, Bsz, Hh, GF_TRANS, 1.f);

  // ---- 4. scores = q k^T / 8 ----
  launch_gemm(stream, 2, qb, (long)Hh * SEQpA * DKv, (long)SEQpA * DKv,
              kb, (long)Hh * SEQpA * DKv, (long)SEQpA * DKv,
              scoresb, (long)Hh * SEQpA * SEQpA * 4, (long)SEQpA * SEQpA * 4,
              nullptr, nullptr, 0, SEQ, SEQ, DKv, DKv, DKv, SEQpA, Bsz, Hh, 0, 0.125f);

  // ---- 5. softmax -> bf16 attn (K padded to 128 with zeros) ----
  {
    long waves = (long)Bsz * Hh * SEQ;
    softmax_kernel<<<(unsigned)((waves + 3) / 4), 128, 0, stream>>>(scoresb, attnb);
  }

  // ---- 6. ctx = attn @ v, written bf16 into concatenated (b,s) x (h*64) ----
  launch_gemm(stream, 1, attnb, (long)Hh * SEQpA * SEQpK, (long)SEQpA * SEQpK,
              vTb, (long)Hh * DKv * SEQpK, (long)DKv * SEQpK,
              ctxb, (long)SEQ * Dp * 2, (long)DKv * 2,
              nullptr, nullptr, 0, SEQ, DKv, SEQpK, SEQpK, SEQpK, Dp, Bsz, Hh, GF_BF16, 1.f);

  // ---- 7. proj = ctx @ wo + wo_b ----
  launch_gemm(stream, 4, ctxb, 0, 0, woT, 0, 0, projb, 0, 0,
              wo_b, nullptr, 0, MROWS, Dm, Dp, Dp, Dp, Dm, 1, 1, GF_BIAS, 1.f);

  // ---- 8. residual + layernorm -> bf16 padded ----
  ln_kernel<<<(MROWS + 3) / 4, 128, 0, stream>>>(projb, data_in, ln_g, ln_b, xlnb);

  // ---- 9. LSTM layer 0 ----
  launch_gemm(stream, 4, xlnb, 0, 0, Wih0b, 0, 0, preb, 0, 0,
              bsum0, nullptr, 0, MROWS, G4D, Dp, Dp, Dp, G4D, 1, 1, GF_BIAS, 1.f);
  fill_bf16_kernel<<<((long)Bsz * Dp + 255) / 256, 256, 0, stream>>>(hbf, (long)Bsz * Dp);
  fill_f32_kernel<<<((long)Bsz * Dm + 255) / 256, 256, 0, stream>>>(cbuf, (long)Bsz * Dm);
  for (int t = 0; t < SEQ; ++t) {
    launch_gemm(stream, 4, hbf, 0, 0, Whh0b, 0, 0, gatesb, 0, 0,
                nullptr, preb + (long)t * G4D, SEQ * G4D,
                Bsz, G4D, Dp, Dp, Dp, G4D, 1, 1, GF_ADD, 1.f);
    lstm_point_kernel<<<((long)Bsz * Dm + 255) / 256, 256, 0, stream>>>(
        gatesb, cbuf, hbf, ys1b, nullptr, t, 1);
  }

  // ---- 10. LSTM layer 1 ----
  launch_gemm(stream, 4, ys1b, 0, 0, Wih1b, 0, 0, preb, 0, 0,
              bsum1, nullptr, 0, MROWS, G4D, Dp, Dp, Dp, G4D, 1, 1, GF_BIAS, 1.f);
  fill_bf16_kernel<<<((long)Bsz * Dp + 255) / 256, 256, 0, stream>>>(hbf, (long)Bsz * Dp);
  fill_f32_kernel<<<((long)Bsz * Dm + 255) / 256, 256, 0, stream>>>(cbuf, (long)Bsz * Dm);
  for (int t = 0; t < SEQ; ++t) {
    launch_gemm(stream, 4, hbf, 0, 0, Whh1b, 0, 0, gatesb, 0, 0,
                nullptr, preb + (long)t * G4D, SEQ * G4D,
                Bsz, G4D, Dp, Dp, Dp, G4D, 1, 1, GF_ADD, 1.f);
    lstm_point_kernel<<<((long)Bsz * Dm + 255) / 256, 256, 0, stream>>>(
        gatesb, cbuf, hbf, nullptr, ys2b, t, 2);
  }

  // ---- 11. sequence-length select + fc1 ----
  lengths_kernel<<<Bsz, 128, 0, stream>>>(data_in, lenb);
  sel_kernel<<<((long)Bsz * Dp + 255) / 256, 256, 0, stream>>>(ys2b, lenb, selb);
  launch_gemm(stream, 2, selb, 0, 0, fc1b, 0, 0, yb, 0, 0,
              fc1_b, nullptr, 0, Bsz, DIM2, Dp, Dp, Dp, DIM2, 1, 1, GF_BIAS, 1.f);

  // ---- 12. batchnorm + leaky relu + fc3 ----
  bnstats_kernel<<<DIM2, 256, 0, stream>>>(yb, bnm, bnv);
  bnfc3_kernel<<<Bsz, 128, 0, stream>>>(yb, bnm, bnv, bn_g, bn_b, fc3_w, fc3_b, out);
}
